// ElasticAttention_90056874262668
// MI455X (gfx1250) — matrix operand order repaired and checked by reading, not compiled, not run on hardware
//
#include <hip/hip_runtime.h>
#include <hip/hip_bf16.h>
#include <math.h>

// ---------------------------------------------------------------------------
// ElasticAttention for MI455X (gfx1250): GQA attention block.
// All matmuls via v_wmma_f32_16x16x32_bf16 (fp32 accumulate), fp32 softmax.
// V tiles staged into LDS with gfx1250 async copies when available.
// ---------------------------------------------------------------------------

typedef __attribute__((ext_vector_type(16))) __bf16 v16bf;
typedef __attribute__((ext_vector_type(8)))  float  v8f;
typedef __attribute__((vector_size(16)))     int    v4i;
typedef __attribute__((address_space(1)))    v4i    as1_v4i;   // global
typedef __attribute__((address_space(3)))    v4i    as3_v4i;   // LDS

#define T_NEW    2048
#define HID      2048
#define N_HEADS  16
#define KV_HEADS 2
#define HEAD_DIM 128
#define T_PAST   2048
#define T_FULL   (T_PAST + T_NEW)
#define KV_N     (KV_HEADS * HEAD_DIM)     // 256
#define SCALE    0.08838834764831845f      // 1/sqrt(HEAD_DIM)
#define NEGBIG   (-1e9f)

#ifndef __has_builtin
#define __has_builtin(x) 0
#endif
#if __has_builtin(__builtin_amdgcn_global_load_async_to_lds_b128)
#define HAVE_ASYNC_LDS 1
#else
#define HAVE_ASYNC_LDS 0
#endif

// RA = reuse-A hint: set only when previous wmma is identical with same A regs.
template <bool RA>
static __device__ __forceinline__ v8f wmma_bf16(v16bf a, v16bf b, v8f c) {
    return __builtin_amdgcn_wmma_f32_16x16x32_bf16(false, a, false, b,
                                                   (short)0, c, RA, false);
}

// A-fragment (16x32 bf16, this lane's half): 16 contiguous at base (K ascending).
static __device__ __forceinline__ v16bf load_a_frag_bf16(const __bf16* base) {
    v16bf a;
#pragma unroll
    for (int i = 0; i < 8; ++i) a[i] = base[i];
#pragma unroll
    for (int i = 0; i < 8; ++i) a[8 + i] = base[8 + i];
    return a;
}

// B-fragment (32x16 bf16, this lane's half): 16 contiguous f32 (K ascending).
static __device__ __forceinline__ v16bf load_b_frag_f32(const float* base) {
    v16bf b;
#pragma unroll
    for (int i = 0; i < 16; ++i) b[i] = (__bf16)base[i];
    return b;
}

// ---------------------------------------------------------------------------
// 0) fp32 -> bf16 pre-convert of hidden_states (done once; reused 160x).
// ---------------------------------------------------------------------------
__global__ void cvt_hidden_kernel(const float4* __restrict__ src,
                                  __bf16* __restrict__ dst) {
    int idx = blockIdx.x * blockDim.x + threadIdx.x;    // 1M threads, 4 elems each
    float4 x = src[idx];
    __bf16* d = dst + (size_t)idx * 4;
    d[0] = (__bf16)x.x; d[1] = (__bf16)x.y;
    d[2] = (__bf16)x.z; d[3] = (__bf16)x.w;
}

// ---------------------------------------------------------------------------
// 1) Copy past_k/past_v into head-major KV cache (rows 0..T_PAST-1).
// ---------------------------------------------------------------------------
__global__ void copy_past_kernel(const float4* __restrict__ pk,
                                 const float4* __restrict__ pv,
                                 float4* __restrict__ kdst,
                                 float4* __restrict__ vdst) {
    const int per = KV_HEADS * T_PAST * HEAD_DIM / 4;   // 131072
    int idx = blockIdx.x * blockDim.x + threadIdx.x;
    if (idx >= 2 * per) return;
    const float4* src; float4* dst; int t = idx;
    if (t < per) { src = pk; dst = kdst; }
    else         { src = pv; dst = vdst; t -= per; }
    int flat = t * 4;
    int hh  = flat / (T_PAST * HEAD_DIM);
    int rem = flat - hh * (T_PAST * HEAD_DIM);
    dst[(hh * T_FULL * HEAD_DIM + rem) >> 2] = src[t];
}

// ---------------------------------------------------------------------------
// 2) C = X @ W^T + b, bf16 A.  One 16x64 output block per wave (4 N-tiles
//    sharing each A fragment, reuse_a set on wmma 2..4), K-loop step 32.
//    MODE 0: bf16 out [M,N] (Q)   MODE 1: f32 KV-cache tail   MODE 2: f32 out
// ---------------------------------------------------------------------------
template <int MODE>
__global__ __launch_bounds__(256, 1)
void gemm_xwT_kernel(const __bf16* __restrict__ A,    // [T_NEW, HID]
                     const float* __restrict__ W,     // [N, HID]
                     const float* __restrict__ bias,  // [N]
                     __bf16* __restrict__ bout,
                     float*  __restrict__ fout,
                     int N) {
    const int lane = threadIdx.x & 31;
    const int gw   = blockIdx.x * (blockDim.x >> 5) + (threadIdx.x >> 5);
    const int tilesN4 = N >> 6;
    const int tm  = gw / tilesN4;
    const int tn4 = gw - tm * tilesN4;
    const int half = lane >> 4;
    const int l16  = lane & 15;

    const __bf16* arow = A + (size_t)(tm * 16 + l16) * HID;
    const float* wrow[4];
#pragma unroll
    for (int j = 0; j < 4; ++j)
        wrow[j] = W + (size_t)(tn4 * 64 + j * 16 + l16) * HID;

    v8f c[4];
#pragma unroll
    for (int j = 0; j < 4; ++j) c[j] = (v8f){};

    for (int k0 = 0; k0 < HID; k0 += 32) {
        v16bf af = load_a_frag_bf16(arow + k0 + half * 16);
        c[0] = wmma_bf16<false>(af, load_b_frag_f32(wrow[0] + k0 + half * 16), c[0]);
#pragma unroll
        for (int j = 1; j < 4; ++j)
            c[j] = wmma_bf16<true>(af, load_b_frag_f32(wrow[j] + k0 + half * 16), c[j]);
    }

#pragma unroll
    for (int j = 0; j < 4; ++j) {
        const int n  = tn4 * 64 + j * 16 + l16;
        const float bv = bias[n];
        if (MODE == 0) {
#pragma unroll
            for (int v = 0; v < 8; ++v) {
                int m = tm * 16 + v + half * 8;
                bout[(size_t)m * N + n] = (__bf16)(c[j][v] + bv);
            }
        } else if (MODE == 1) {
            int h = n >> 7, d = n & 127;
            float* dst = fout + (size_t)h * T_FULL * HEAD_DIM
                              + (size_t)T_PAST * HEAD_DIM + d;
#pragma unroll
            for (int v = 0; v < 8; ++v) {
                int m = tm * 16 + v + half * 8;
                dst[(size_t)m * HEAD_DIM] = c[j][v] + bv;
            }
        } else {
#pragma unroll
            for (int v = 0; v < 8; ++v) {
                int m = tm * 16 + v + half * 8;
                fout[(size_t)m * N + n] = c[j][v] + bv;
            }
        }
    }
}

// ---------------------------------------------------------------------------
// 3) Flash attention: one wave per (head, 16-query-row tile).
//    S = Q K^T * scale + causal_mask ; online softmax ; O = P V.
//    V tiles staged to LDS via async copy engine (ASYNCcnt) when available.
// ---------------------------------------------------------------------------
__global__ __launch_bounds__(256, 1)
void attn_kernel(const __bf16* __restrict__ qbuf,   // [T_NEW, HID]
                 const float*  __restrict__ kcache, // [KVH, T_FULL, D]
                 const float*  __restrict__ vcache,
                 __bf16* __restrict__ obuf) {       // [T_NEW, HID]
    __shared__ __bf16 pshare[8][16 * 32];                      // per-wave P tile
#if HAVE_ASYNC_LDS
    __shared__ float  vshare[8][32 * HEAD_DIM];                // 16KB per wave
#endif
    const int lane = threadIdx.x & 31;
    const int wv   = threadIdx.x >> 5;
    const int gw   = blockIdx.x * (blockDim.x >> 5) + wv;
    const int h    = gw >> 7;                 // head 0..15
    const int qt   = gw & 127;                // query tile 0..127
    const int q0   = qt * 16;
    const int kvh  = h / (N_HEADS / KV_HEADS);
    const int half = lane >> 4;
    const int l16  = lane & 15;

    // Q fragments for all 4 K-steps of the 128-wide head dim.
    v16bf qa[4];
    {
        const __bf16* qrow = qbuf + (size_t)(q0 + l16) * HID + h * HEAD_DIM;
#pragma unroll
        for (int ks = 0; ks < 4; ++ks)
            qa[ks] = load_a_frag_bf16(qrow + ks * 32 + half * 16);
    }

    const float* kbase = kcache + (size_t)kvh * T_FULL * HEAD_DIM;
    const float* vbase = vcache + (size_t)kvh * T_FULL * HEAD_DIM;

    v8f o[8];
#pragma unroll
    for (int j = 0; j < 8; ++j) o[j] = (v8f){};
    float rmax[8], rsum[8];
#pragma unroll
    for (int v = 0; v < 8; ++v) { rmax[v] = -3.0e38f; rsum[v] = 0.0f; }

    const int kv_len = T_PAST + q0 + 16;      // causal: positions [0, kv_len)
    for (int k0 = 0; k0 < kv_len; k0 += 32) {
#if HAVE_ASYNC_LDS
        // Kick off async V-tile copy (32 rows x 512B; one row per issue,
        // 16B per lane) so it overlaps the S-tile WMMA work below.
        {
            float* ldst = &vshare[wv][0] + lane * 4;
            const float* gsrc = vbase + (size_t)k0 * HEAD_DIM + lane * 4;
#pragma unroll
            for (int r = 0; r < 32; ++r)
                __builtin_amdgcn_global_load_async_to_lds_b128(
                    (as1_v4i*)(gsrc + (size_t)r * HEAD_DIM),
                    (as3_v4i*)(ldst + r * HEAD_DIM), 0, 0);
        }
#endif
        if (k0 + 32 < kv_len)                 // prefetch next K tile
            __builtin_prefetch(kbase + (size_t)(k0 + 32 + l16) * HEAD_DIM, 0, 1);

        // ---- S tile (16 x 32) = Q @ K^T ----
        v8f s[2]; s[0] = (v8f){}; s[1] = (v8f){};
#pragma unroll
        for (int sub = 0; sub < 2; ++sub) {
            const float* krow = kbase + (size_t)(k0 + sub * 16 + l16) * HEAD_DIM
                                      + half * 16;
#pragma unroll
            for (int ks = 0; ks < 4; ++ks)
                s[sub] = wmma_bf16<false>(qa[ks],
                                          load_b_frag_f32(krow + ks * 32), s[sub]);
        }

        // ---- scale + causal mask + row max (reduce across 16-lane half) ----
        float pmax[8];
#pragma unroll
        for (int v = 0; v < 8; ++v) {
            int lim = T_PAST + q0 + v + half * 8;     // max allowed k position
            float s0 = s[0][v] * SCALE;
            float s1 = s[1][v] * SCALE;
            if (k0 + l16      > lim) s0 += NEGBIG;
            if (k0 + 16 + l16 > lim) s1 += NEGBIG;
            s[0][v] = s0; s[1][v] = s1;
            float mv = fmaxf(s0, s1);
            mv = fmaxf(mv, __shfl_xor(mv, 1, 32));
            mv = fmaxf(mv, __shfl_xor(mv, 2, 32));
            mv = fmaxf(mv, __shfl_xor(mv, 4, 32));
            mv = fmaxf(mv, __shfl_xor(mv, 8, 32));
            pmax[v] = mv;
        }

        // ---- online softmax update ----
#pragma unroll
        for (int v = 0; v < 8; ++v) {
            float nm    = fmaxf(rmax[v], pmax[v]);
            float alpha = __expf(rmax[v] - nm);
            rmax[v] = nm;
            float p0 = __expf(s[0][v] - nm);
            float p1 = __expf(s[1][v] - nm);
            s[0][v] = p0; s[1][v] = p1;
            float ps = p0 + p1;
            ps += __shfl_xor(ps, 1, 32);
            ps += __shfl_xor(ps, 2, 32);
            ps += __shfl_xor(ps, 4, 32);
            ps += __shfl_xor(ps, 8, 32);
            rsum[v] = rsum[v] * alpha + ps;
#pragma unroll
            for (int j = 0; j < 8; ++j) o[j][v] *= alpha;
        }

        // ---- P: C-layout -> A-layout via per-wave LDS bounce ----
        __bf16* pb = &pshare[wv][0];
#pragma unroll
        for (int v = 0; v < 8; ++v) {
            int m = v + half * 8;
            pb[m * 32 + l16]      = (__bf16)s[0][v];
            pb[m * 32 + 16 + l16] = (__bf16)s[1][v];
        }
        v16bf pf = load_a_frag_bf16(pb + l16 * 32 + half * 16);

#if HAVE_ASYNC_LDS
#if __has_builtin(__builtin_amdgcn_s_wait_asynccnt)
        __builtin_amdgcn_s_wait_asynccnt(0);
#else
        asm volatile("s_wait_asynccnt 0" ::: "memory");
#endif
#endif
        // ---- O (16 x 128) += P (16x32) @ V (32x128), reuse_a on j>0 ----
#pragma unroll
        for (int j = 0; j < 8; ++j) {
            v16bf vf;
#if HAVE_ASYNC_LDS
            const float* vcol = &vshare[wv][(half * 16) * HEAD_DIM + j * 16 + l16];
#else
            const float* vcol = vbase + (size_t)(k0 + half * 16) * HEAD_DIM
                                      + j * 16 + l16;
#endif
#pragma unroll
            for (int kk = 0; kk < 16; ++kk)
                vf[kk] = (__bf16)vcol[(size_t)kk * HEAD_DIM];
            if (j == 0) o[j] = wmma_bf16<false>(pf, vf, o[j]);
            else        o[j] = wmma_bf16<true >(pf, vf, o[j]);
        }
    }

    // ---- normalize + store bf16 ----
    const int n0 = h * HEAD_DIM;
#pragma unroll
    for (int v = 0; v < 8; ++v) {
        float inv = 1.0f / rsum[v];
        int m = q0 + v + half * 8;
#pragma unroll
        for (int j = 0; j < 8; ++j)
            obuf[(size_t)m * HID + n0 + j * 16 + l16] = (__bf16)(o[j][v] * inv);
    }
}

// ---------------------------------------------------------------------------
extern "C" void kernel_launch(void* const* d_in, const int* in_sizes, int n_in,
                              void* d_out, int out_size, void* d_ws, size_t ws_size,
                              hipStream_t stream) {
    const float* hidden = (const float*)d_in[0];
    /* d_in[1] = attention_mask: causal mask computed analytically, unused */
    const float* past_k = (const float*)d_in[2];
    const float* past_v = (const float*)d_in[3];
    const float* q_w = (const float*)d_in[4];
    const float* q_b = (const float*)d_in[5];
    const float* k_w = (const float*)d_in[6];
    const float* k_b = (const float*)d_in[7];
    const float* v_w = (const float*)d_in[8];
    const float* v_b = (const float*)d_in[9];
    const float* o_w = (const float*)d_in[10];
    const float* o_b = (const float*)d_in[11];

    // d_out = [out 4194304][new_k 1048576][new_v 1048576] floats
    float* out    = (float*)d_out;
    float* kcache = out + (size_t)T_NEW * HID;
    float* vcache = kcache + (size_t)KV_HEADS * T_FULL * HEAD_DIM;

    // workspace (16MB): [hbuf bf16 8MB | qbuf bf16 8MB]
    // hbuf (bf16 hidden) is dead after the QKV GEMMs, so the attention
    // output buffer aliases it.
    __bf16* hbuf = (__bf16*)d_ws;
    __bf16* qbuf = (__bf16*)((char*)d_ws + (size_t)T_NEW * HID * 2);
    __bf16* obuf = hbuf;

    // 0) hidden fp32 -> bf16 once
    cvt_hidden_kernel<<<4096, 256, 0, stream>>>((const float4*)hidden, hbuf);
    // 1) cache concat (past rows)
    copy_past_kernel<<<1024, 256, 0, stream>>>((const float4*)past_k,
                                               (const float4*)past_v,
                                               (float4*)kcache, (float4*)vcache);
    // 2) projections: Q (128x32 blocks of 16x64), K and V (128x4 blocks)
    gemm_xwT_kernel<0><<<512, 256, 0, stream>>>(hbuf, q_w, q_b, qbuf, nullptr, HID);
    gemm_xwT_kernel<1><<<64,  256, 0, stream>>>(hbuf, k_w, k_b, nullptr, kcache, KV_N);
    gemm_xwT_kernel<1><<<64,  256, 0, stream>>>(hbuf, v_w, v_b, nullptr, vcache, KV_N);
    // 3) attention: 16 heads x 128 query tiles = 2048 waves
    attn_kernel<<<256, 256, 0, stream>>>(qbuf, kcache, vcache, obuf);
    // 4) output projection
    gemm_xwT_kernel<2><<<512, 256, 0, stream>>>(obuf, o_w, o_b, nullptr, out, HID);
}
